// DecomposableAttention_240518168600
// MI455X (gfx1250) — matrix operand order repaired and checked by reading, not compiled, not run on hardware
//
#include <hip/hip_runtime.h>
#include <hip/hip_bf16.h>

typedef _Float16 v16h __attribute__((ext_vector_type(16)));
typedef float    v8f  __attribute__((ext_vector_type(8)));
typedef _Float16 f16x2 __attribute__((ext_vector_type(2)));

enum { EPI_NONE = 0, EPI_BIASRELU = 1, EPI_EXP = 2, EPI_ROWSCALE = 3 };

#define BM   128
#define BN   128
#define BKS  32
#define LSTR 34   // 34 halves = 68B row stride = 17 banks, gcd(17,64)=1 -> conflict-free

// ---------------------------------------------------------------------------
// Stage one 128x32 A tile and one 128x32 B tile (f32 -> f16) into LDS.
//   A logical MxK. ATRANS=0: A row-major MxK. ATRANS=1: A stored KxM.
//   B logical KxN. BTRANS=0: B row-major KxN. BTRANS=1: B stored NxK.
// LDS layouts: As[m][k] (k contiguous), Bs[n][k] (k contiguous).
// ---------------------------------------------------------------------------
template <bool ATRANS, bool BTRANS>
__device__ __forceinline__ void stage_tiles(
    _Float16* __restrict__ As, _Float16* __restrict__ Bs,
    const float* __restrict__ Ab, const float* __restrict__ Bb,
    int m0, int n0, int k0, int lda, int ldb, int tid)
{
    #pragma unroll
    for (int r = 0; r < 32; ++r) {
        int idx = r * 128 + tid;
        if (!ATRANS) {
            int row = idx >> 5, col = idx & 31;              // tid -> contiguous k
            As[row * LSTR + col] =
                (_Float16)Ab[(size_t)(m0 + row) * lda + (k0 + col)];
        } else {
            int col = idx >> 7, row = idx & 127;             // tid -> contiguous m
            As[row * LSTR + col] =
                (_Float16)Ab[(size_t)(k0 + col) * lda + (m0 + row)];
        }
    }
    #pragma unroll
    for (int r = 0; r < 32; ++r) {
        int idx = r * 128 + tid;
        if (!BTRANS) {                                       // B is KxN
            int col = idx >> 7, n = idx & 127;               // tid -> contiguous n
            Bs[n * LSTR + col] =
                (_Float16)Bb[(size_t)(k0 + col) * ldb + (n0 + n)];
        } else {                                             // B is NxK
            int n = idx >> 5, col = idx & 31;                // tid -> contiguous k
            Bs[n * LSTR + col] =
                (_Float16)Bb[(size_t)(n0 + n) * ldb + (k0 + col)];
        }
    }
}

// ---------------------------------------------------------------------------
// Tiled wave32 WMMA GEMM: C = epilogue(op(A) @ op(B))
// Block: 128 threads = 4 waves (2x2); block tile 128x128; wave tile 64x64
// = 4x4 v_wmma_f32_16x16x32_f16 tiles. Double-buffered LDS, one barrier
// per 32-K step; 16 WMMAs per 8 fragment loads per wave per step.
// grid = (N/128, M/128, batches); per-batch strides bA/bB/bC/bS.
// ---------------------------------------------------------------------------
template <bool ATRANS, bool BTRANS, int EPI>
__global__ __launch_bounds__(128, 1)
void wmma_gemm(const float* __restrict__ A, const float* __restrict__ Bm,
               const float* __restrict__ bias, const float* __restrict__ rscale,
               float* __restrict__ C, int K,
               int lda, int ldb, int ldc,
               long long bA, long long bB, long long bC, long long bS)
{
    __shared__ _Float16 As[2][BM * LSTR];
    __shared__ _Float16 Bs[2][BN * LSTR];

    const float* Ab = A  + (long long)blockIdx.z * bA;
    const float* Bb = Bm + (long long)blockIdx.z * bB;
    float*       Cb = C  + (long long)blockIdx.z * bC;

    const int m0   = blockIdx.y * BM;
    const int n0   = blockIdx.x * BN;
    const int tid  = threadIdx.x;
    const int lane = tid & 31;
    const int wv   = tid >> 5;
    const int wM   = (wv >> 1) * 64;   // wave row offset in block tile
    const int wN   = (wv & 1) * 64;    // wave col offset in block tile
    const int nl   = lane & 15;        // lane % 16
    const int hi   = lane >> 4;        // lane half

    v8f acc[4][4] = {};

    stage_tiles<ATRANS, BTRANS>(As[0], Bs[0], Ab, Bb, m0, n0, 0, lda, ldb, tid);
    __syncthreads();

    const int nsteps = K >> 5;
    for (int s = 0; s < nsteps; ++s) {
        const int cur = s & 1;
        // prefetch next tile into the other buffer while we compute this one
        if (s + 1 < nsteps)
            stage_tiles<ATRANS, BTRANS>(As[cur ^ 1], Bs[cur ^ 1], Ab, Bb,
                                        m0, n0, (s + 1) << 5, lda, ldb, tid);

        // ---- A fragments 16x32: lane (m = nl, half = hi);
        //      elem e: K = (e>=8?16:0) + hi*8 + (e%8) ----
        v16h af[4];
        #pragma unroll
        for (int mt = 0; mt < 4; ++mt) {
            const _Float16* ap = &As[cur][(wM + mt * 16 + nl) * LSTR];
            v16h a;
            #pragma unroll
            for (int v = 0; v < 8; ++v) {
                int kp = ((v & 4) << 2) + hi * 8 + ((v & 3) << 1);
                f16x2 p = *(const f16x2*)(ap + kp);
                a[2 * v]     = p.x;
                a[2 * v + 1] = p.y;
            }
            af[mt] = a;
        }
        // ---- stream B fragments (32x16: lane n = nl; elem e: K = (e>=8?16:0) + hi*8 + (e%8)) ----
        #pragma unroll
        for (int nt = 0; nt < 4; ++nt) {
            const _Float16* bp = &Bs[cur][(wN + nt * 16 + nl) * LSTR];
            v16h b;
            #pragma unroll
            for (int v = 0; v < 8; ++v) {
                int kp = ((v & 4) << 2) + hi * 8 + ((v & 3) << 1);
                f16x2 p = *(const f16x2*)(bp + kp);
                b[2 * v]     = p.x;
                b[2 * v + 1] = p.y;
            }
            #pragma unroll
            for (int mt = 0; mt < 4; ++mt)
                acc[mt][nt] = __builtin_amdgcn_wmma_f32_16x16x32_f16(
                    false, af[mt], false, b, (short)0, acc[mt][nt],
                    false, false);
        }
        __syncthreads();
    }

    // ---- epilogue + writeback; C/D layout: n = lane%16, m = v + (lane/16)*8 ----
    #pragma unroll
    for (int mt = 0; mt < 4; ++mt) {
        #pragma unroll
        for (int nt = 0; nt < 4; ++nt) {
            const int nG    = n0 + wN + nt * 16 + nl;
            const int mBase = m0 + wM + mt * 16 + hi * 8;
            #pragma unroll
            for (int v = 0; v < 8; ++v) {
                int   m   = mBase + v;
                float val = acc[mt][nt][v];
                if (EPI == EPI_BIASRELU)      val = fmaxf(val + bias[nG], 0.0f);
                else if (EPI == EPI_EXP)      val = __expf(val);
                else if (EPI == EPI_ROWSCALE) val = val * (rscale + (long long)blockIdx.z * bS)[m];
                Cb[(size_t)m * ldc + nG] = val;
            }
        }
    }
}

// invRowSum[b][i] = 1 / sum_j e[b][i][j]      (grid = (L, B), block = 256)
__global__ void rowsum_inv_kernel(const float* __restrict__ e, float* __restrict__ out, int L)
{
    int b = blockIdx.y, i = blockIdx.x;
    const float* row = e + ((size_t)b * L + i) * L;
    float s = 0.f;
    for (int j = threadIdx.x; j < L; j += 256) s += row[j];
    __shared__ float red[256];
    red[threadIdx.x] = s;
    __syncthreads();
    for (int off = 128; off > 0; off >>= 1) {
        if (threadIdx.x < off) red[threadIdx.x] += red[threadIdx.x + off];
        __syncthreads();
    }
    if (threadIdx.x == 0) out[(size_t)b * L + i] = 1.0f / red[0];
}

// invColSum[b][j] = 1 / sum_i e[b][i][j]      (grid = (L/256, B), block = 256)
__global__ void colsum_inv_kernel(const float* __restrict__ e, float* __restrict__ out, int L)
{
    int b = blockIdx.y;
    int j = blockIdx.x * blockDim.x + threadIdx.x;
    if (j >= L) return;
    const float* eb = e + (size_t)b * L * L;
    float s = 0.f;
    for (int i = 0; i < L; ++i) s += eb[(size_t)i * L + j];
    out[(size_t)b * L + j] = 1.0f / s;
}

// agg[b][off + h] += sum_i p[b][i][h]         (grid = (H/256, B), block = 256)
__global__ void col_accum_add(const float* __restrict__ p, float* __restrict__ agg,
                              int L, int H, int off)
{
    int b = blockIdx.y;
    int h = blockIdx.x * blockDim.x + threadIdx.x;
    if (h >= H) return;
    const float* pb = p + (size_t)b * L * H;
    float s = 0.f;
    for (int i = 0; i < L; ++i) s += pb[(size_t)i * H + h];
    agg[(size_t)b * (2 * H) + off + h] += s;
}

__global__ void zero_kernel(float* __restrict__ p, int n)
{
    int i = blockIdx.x * blockDim.x + threadIdx.x;
    if (i < n) p[i] = 0.0f;
}

// tiny GEMM for the h-MLP (M = 2 rows): C = relu(A @ W + b)
__global__ void small_gemm_bias_relu(const float* __restrict__ A, const float* __restrict__ W,
                                     const float* __restrict__ bias, float* __restrict__ C,
                                     int M, int N, int K)
{
    int n = blockIdx.x * blockDim.x + threadIdx.x;
    if (n >= N) return;
    for (int m = 0; m < M; ++m) {
        float acc = 0.f;
        for (int k = 0; k < K; ++k) acc += A[(size_t)m * K + k] * W[(size_t)k * N + n];
        C[(size_t)m * N + n] = fmaxf(acc + bias[n], 0.0f);
    }
}

extern "C" void kernel_launch(void* const* d_in, const int* in_sizes, int n_in,
                              void* d_out, int out_size, void* d_ws, size_t ws_size,
                              hipStream_t stream)
{
    (void)in_sizes; (void)n_in; (void)out_size; (void)ws_size;

    const float* x1  = (const float*)d_in[0];
    const float* x2  = (const float*)d_in[1];
    const float* fW1 = (const float*)d_in[2];  const float* fb1 = (const float*)d_in[3];
    const float* fW2 = (const float*)d_in[4];  const float* fb2 = (const float*)d_in[5];
    const float* fW3 = (const float*)d_in[6];  const float* fb3 = (const float*)d_in[7];
    const float* gW1 = (const float*)d_in[8];  const float* gb1 = (const float*)d_in[9];
    const float* gW2 = (const float*)d_in[10]; const float* gb2 = (const float*)d_in[11];
    const float* gW3 = (const float*)d_in[12]; const float* gb3 = (const float*)d_in[13];
    const float* hW1 = (const float*)d_in[14]; const float* hb1 = (const float*)d_in[15];
    const float* hW2 = (const float*)d_in[16]; const float* hb2 = (const float*)d_in[17];
    const float* hW3 = (const float*)d_in[18]; const float* hb3 = (const float*)d_in[19];

    constexpr int B = 2, L = 4096, E = 1024, H = 2048, O = 3;
    constexpr int BL = B * L;  // 8192 rows (batches contiguous)

    // ---- workspace carve-out ----
    float* w = (float*)d_ws;
    size_t off = 0;
    auto alloc = [&](size_t n) { float* p = w + off; off += n; return p; };
    float* tmp1   = alloc((size_t)BL * H);        // 8192 x 2048
    float* tmp2   = alloc((size_t)BL * H);
    float* fa     = alloc((size_t)BL * H);
    float* fbuf   = alloc((size_t)BL * H);
    float* e      = alloc((size_t)B * L * L);     // 2 x 4096 x 4096
    float* invRow = alloc((size_t)B * L);
    float* invCol = alloc((size_t)B * L);
    float* beta   = alloc((size_t)B * L * E);
    float* alpha  = alloc((size_t)B * L * E);
    float* agg    = alloc((size_t)B * 2 * H);     // 2 x 4096
    float* t1     = alloc((size_t)B * H);
    float* t2     = alloc((size_t)B * H);

    const dim3 blk(128);
    auto grid2 = [](int M, int N, int Z) { return dim3(N / BN, M / BM, Z); };

    // ---------------- Attend: fa = f(x1), fb = f(x2) ----------------
    wmma_gemm<false, false, EPI_BIASRELU><<<grid2(BL, H, 1), blk, 0, stream>>>(
        x1, fW1, fb1, nullptr, tmp1, E, E, H, H, 0, 0, 0, 0);
    wmma_gemm<false, false, EPI_BIASRELU><<<grid2(BL, H, 1), blk, 0, stream>>>(
        tmp1, fW2, fb2, nullptr, tmp2, H, H, H, H, 0, 0, 0, 0);
    wmma_gemm<false, false, EPI_BIASRELU><<<grid2(BL, H, 1), blk, 0, stream>>>(
        tmp2, fW3, fb3, nullptr, fa, H, H, H, H, 0, 0, 0, 0);

    wmma_gemm<false, false, EPI_BIASRELU><<<grid2(BL, H, 1), blk, 0, stream>>>(
        x2, fW1, fb1, nullptr, tmp1, E, E, H, H, 0, 0, 0, 0);
    wmma_gemm<false, false, EPI_BIASRELU><<<grid2(BL, H, 1), blk, 0, stream>>>(
        tmp1, fW2, fb2, nullptr, tmp2, H, H, H, H, 0, 0, 0, 0);
    wmma_gemm<false, false, EPI_BIASRELU><<<grid2(BL, H, 1), blk, 0, stream>>>(
        tmp2, fW3, fb3, nullptr, fbuf, H, H, H, H, 0, 0, 0, 0);

    // ---------------- e = exp(fa @ fb^T), per batch ----------------
    wmma_gemm<false, true, EPI_EXP><<<grid2(L, L, B), blk, 0, stream>>>(
        fa, fbuf, nullptr, nullptr, e, H, H, H, L,
        (long long)L * H, (long long)L * H, (long long)L * L, 0);

    // ---------------- normalization sums ----------------
    rowsum_inv_kernel<<<dim3(L, B), 256, 0, stream>>>(e, invRow, L);
    colsum_inv_kernel<<<dim3(L / 256, B), 256, 0, stream>>>(e, invCol, L);

    // beta[b]  = diag(invRow[b]) @ e[b]   @ x2[b]   (M=L, N=E, K=L)
    wmma_gemm<false, false, EPI_ROWSCALE><<<grid2(L, E, B), blk, 0, stream>>>(
        e, x2, nullptr, invRow, beta, L, L, E, E,
        (long long)L * L, (long long)L * E, (long long)L * E, (long long)L);
    // alpha[b] = diag(invCol[b]) @ e[b]^T @ x1[b]
    wmma_gemm<true, false, EPI_ROWSCALE><<<grid2(L, E, B), blk, 0, stream>>>(
        e, x1, nullptr, invCol, alpha, L, L, E, E,
        (long long)L * L, (long long)L * E, (long long)L * E, (long long)L);

    // ---------------- Compare + Aggregate ----------------
    // concat along seq dim then sum over seq  ==  sum(g(part1)) + sum(g(part2))
    zero_kernel<<<(B * 2 * H + 255) / 256, 256, 0, stream>>>(agg, B * 2 * H);

    const float* gsrc[4] = { x1, beta, x2, alpha };
    const int    goff[4] = { 0, 0, H, H };
    for (int s = 0; s < 4; ++s) {
        wmma_gemm<false, false, EPI_BIASRELU><<<grid2(BL, H, 1), blk, 0, stream>>>(
            gsrc[s], gW1, gb1, nullptr, tmp1, E, E, H, H, 0, 0, 0, 0);
        wmma_gemm<false, false, EPI_BIASRELU><<<grid2(BL, H, 1), blk, 0, stream>>>(
            tmp1, gW2, gb2, nullptr, tmp2, H, H, H, H, 0, 0, 0, 0);
        wmma_gemm<false, false, EPI_BIASRELU><<<grid2(BL, H, 1), blk, 0, stream>>>(
            tmp2, gW3, gb3, nullptr, tmp1, H, H, H, H, 0, 0, 0, 0);
        col_accum_add<<<dim3(H / 256, B), 256, 0, stream>>>(tmp1, agg, L, H, goff[s]);
    }

    // ---------------- h-MLP on (2 x 4096) ----------------
    small_gemm_bias_relu<<<dim3((H + 63) / 64), 64, 0, stream>>>(agg, hW1, hb1, t1, B, H, 2 * H);
    small_gemm_bias_relu<<<dim3((H + 63) / 64), 64, 0, stream>>>(t1, hW2, hb2, t2, B, H, H);
    small_gemm_bias_relu<<<dim3(1), 64, 0, stream>>>(t2, hW3, hb3, (float*)d_out, B, O, H);
}
